// CrystalGraphConvNet_33741263078232
// MI455X (gfx1250) — matrix operand order repaired and checked by reading, not compiled, not run on hardware
//
#include <hip/hip_runtime.h>
#include <stdint.h>

// CGCNN forward for MI455X (gfx1250, wave32, WMMA).
// Strategy: f16 WMMA GEMMs with f32 accumulation; two-pass BatchNorm
// (stats pass + fused apply pass) so nothing larger than ~27 MB of scratch
// is needed and all reused operands (af16 5MB, nbr_fea 39MB, packed W 160KB)
// stay resident in the 192MB L2. Linear-layer biases before batch-stat BN
// cancel and are dropped (convs_b / dist_b / const_b).

typedef __attribute__((ext_vector_type(16))) _Float16 v16h;
typedef __attribute__((ext_vector_type(8)))  float    v8f;

#define N_ATOM   20000
#define MNBR     12
#define NBR_FEA  41
#define IN2      297          // 2*128 + 41
#define KPAD     320          // padded K, 10 WMMA steps of 32
#define KSTEPS   10
#define NROWS    (N_ATOM * MNBR)
#define EPS_BN   1e-5f

union ABfrag { uint4 q[2]; v16h v; };

__device__ __forceinline__ float leakyf(float x)   { return x >= 0.f ? x : 0.01f * x; }
__device__ __forceinline__ float sigmoidf_(float x){ return 1.f / (1.f + expf(-x)); }
__device__ __forceinline__ float softplusf_(float x){ return x > 20.f ? x : log1pf(expf(x)); }

// ---------------------------------------------------------------- embedding
__global__ void k_embed(const float* __restrict__ atom_fea,
                        const float* __restrict__ emb_W,
                        const float* __restrict__ emb_b,
                        float* __restrict__ af, _Float16* __restrict__ af16) {
  __shared__ float row[92];
  int n = blockIdx.x, c = threadIdx.x;
  if (c < 92) row[c] = atom_fea[(size_t)n * 92 + c];
  __syncthreads();
  float s = emb_b[c];
  #pragma unroll 4
  for (int k = 0; k < 92; ++k) s += emb_W[c * 92 + k] * row[k];
  af[(size_t)n * 128 + c]   = s;
  af16[(size_t)n * 128 + c] = (_Float16)s;
}

// ---------------------------------------------------------------- PBC distance
__global__ void k_distance(const float* __restrict__ pos,
                           const float* __restrict__ off,
                           const float* __restrict__ cells,
                           const int* __restrict__ idx,
                           float* __restrict__ dist) {
  int g = blockIdx.x * blockDim.x + threadIdx.x;
  if (g >= NROWS) return;
  int n = g / MNBR;
  int j = idx[g];
  const float* o = off + (size_t)g * 3;
  float dsum = 1e-12f;
  #pragma unroll
  for (int d = 0; d < 3; ++d) {
    float ab = pos[(size_t)j * 3 + d] + o[0] * cells[d] + o[1] * cells[3 + d] + o[2] * cells[6 + d];
    float dv = ab - pos[(size_t)n * 3 + d];
    dsum += dv * dv;
  }
  dist[g] = sqrtf(dsum);
}

// ------------------------------------------- pack weights into B-fragment order
// wp[((c*KSTEPS + k)*32 + lane)*16 + e] = W[col][K], col = c*16 + (lane&15),
// K = k*32 + (lane>>4)*16 + e  (zero-padded past 297). Contiguous k order per
// lane half (same as the A loads), so the GEMM does 32B contiguous loads per lane.
__global__ void k_wpack(const float* __restrict__ W0, const float* __restrict__ W1,
                        int split, _Float16* __restrict__ wp) {
  int g = blockIdx.x * 256 + threadIdx.x;           // 81920 total
  int e = g & 15, lane = (g >> 4) & 31, k = (g >> 9) % KSTEPS, c = g / (KSTEPS * 512);
  int col = c * 16 + (lane & 15);
  int K   = k * 32 + (lane >> 4) * 16 + e;
  float v = 0.f;
  if (K < IN2) v = (col < split) ? W0[(size_t)col * IN2 + K]
                                 : W1[(size_t)(col - split) * IN2 + K];
  wp[g] = (_Float16)v;
}

// ---------------------------------- stage 48 gathered rows (4 atoms) into LDS
__device__ __forceinline__ void fill_X(_Float16* X, const _Float16* __restrict__ af16,
                                       const float* __restrict__ nbr_fea,
                                       const int* __restrict__ nbr_idx,
                                       int n0, int tid) {
  int r = tid % 48, half = tid / 48;                // 2 threads per row
  int n = n0 + r / MNBR, m = r % MNBR;
  int j = nbr_idx[(size_t)n * MNBR + m];
  _Float16* row = X + (size_t)r * KPAD;
  uint32_t* dst = (uint32_t*)row;
  if (half == 0) {                                  // cols 0..159
    const uint32_t* s0 = (const uint32_t*)(af16 + (size_t)n * 128);
    #pragma unroll
    for (int i = 0; i < 64; ++i) dst[i] = s0[i];
    const uint32_t* s1 = (const uint32_t*)(af16 + (size_t)j * 128);
    #pragma unroll
    for (int i = 0; i < 16; ++i) dst[64 + i] = s1[i];
  } else {                                          // cols 160..319
    const uint32_t* s1 = (const uint32_t*)(af16 + (size_t)j * 128);
    #pragma unroll
    for (int i = 16; i < 64; ++i) dst[64 + i] = s1[i];
    const float* nf = nbr_fea + (size_t)(n * MNBR + m) * NBR_FEA;
    #pragma unroll
    for (int c = 0; c < NBR_FEA; ++c) row[256 + c] = (_Float16)nf[c];
    #pragma unroll
    for (int c = IN2; c < KPAD; ++c) row[c] = (_Float16)0.f;
  }
}

// ---------------------------------- core WMMA GEMM: one wave, 16 rows x 256 cols
// A layout (16-bit 16x32): lane row = lane&15, half = lane>>4 selects
// K block [16h..16h+15] (matches packed B) -> two 16B LDS loads.
__device__ __forceinline__ void gemm48(const _Float16* X, const _Float16* __restrict__ wp,
                                       int lane, int wave, v8f acc[16]) {
  int hh = lane >> 4;
  int rrow = wave * 16 + (lane & 15);
  const uint4* arow = (const uint4*)(X + (size_t)rrow * KPAD);
  #pragma unroll
  for (int k = 0; k < KSTEPS; ++k) {
    ABfrag a;
    int f0 = k * 32 + 16 * hh;                // f16 index of chunk0 (16B aligned)
    a.q[0] = arow[f0 >> 3];
    a.q[1] = arow[(f0 >> 3) + 1];             // +8 f16
    __builtin_prefetch(wp + ((size_t)k + 1) * 512, 0, 0);  // hint next K slice
    #pragma unroll
    for (int t = 0; t < 16; ++t) {
      ABfrag b;
      const uint4* bp = (const uint4*)(wp + (((size_t)t * KSTEPS + k) * 32 + lane) * 16);
      b.q[0] = bp[0];
      b.q[1] = bp[1];
      acc[t] = __builtin_amdgcn_wmma_f32_16x16x32_f16(false, a.v, false, b.v,
                                                      (short)0, acc[t], false, false);
    }
  }
}

// ------------------------------- pass 1: GEMM + per-column sum / sumsq (BN stats)
__global__ __launch_bounds__(96) void k_gemm_stats(
    const _Float16* __restrict__ af16, const float* __restrict__ nbr_fea,
    const int* __restrict__ nbr_idx, const _Float16* __restrict__ wp,
    float* __restrict__ gstats) {
  __shared__ __align__(16) _Float16 X[48 * KPAD];
  __shared__ float lst[512];                  // [0..255]=sum, [256..511]=sumsq
  int tid = threadIdx.x, n0 = blockIdx.x * 4;
  for (int c = tid; c < 512; c += 96) lst[c] = 0.f;
  fill_X(X, af16, nbr_fea, nbr_idx, n0, tid);
  __syncthreads();

  int lane = tid & 31, wave = tid >> 5;
  v8f acc[16]; v8f zero = {};
  #pragma unroll
  for (int t = 0; t < 16; ++t) acc[t] = zero;
  gemm48(X, wp, lane, wave, acc);

  int col16 = lane & 15;
  #pragma unroll
  for (int t = 0; t < 16; ++t) {
    float s = 0.f, q = 0.f;
    #pragma unroll
    for (int j2 = 0; j2 < 8; ++j2) { float v = acc[t][j2]; s += v; q += v * v; }
    atomicAdd(&lst[t * 16 + col16], s);
    atomicAdd(&lst[256 + t * 16 + col16], q);
  }
  __syncthreads();
  for (int c = tid; c < 512; c += 96) atomicAdd(&gstats[c], lst[c]);
}

// ------------------------------- fold BN stats + gamma/beta into scale/shift
__global__ void k_finalize(const float* __restrict__ gstats, float cnt,
                           const float* __restrict__ g0, const float* __restrict__ b0,
                           const float* __restrict__ g1, const float* __restrict__ b1,
                           int split, float* __restrict__ sc, float* __restrict__ sh) {
  int c = threadIdx.x;
  if (c >= 256) return;
  float mean = gstats[c] / cnt;
  float var  = gstats[256 + c] / cnt - mean * mean;
  float rstd = rsqrtf(var + EPS_BN);
  float gm = c < split ? g0[c] : g1[c - split];
  float bt = c < split ? b0[c] : b1[c - split];
  float s = gm * rstd;
  sc[c] = s;
  sh[c] = bt - mean * s;
}

// ------------------------------- pass 2 (conv): GEMM + BN + sigmoid*leaky + sum_m
__global__ __launch_bounds__(96) void k_conv_apply(
    const _Float16* __restrict__ af16, const float* __restrict__ nbr_fea,
    const int* __restrict__ nbr_idx, const _Float16* __restrict__ wp,
    const float* __restrict__ sc, const float* __restrict__ sh,
    float* __restrict__ S) {
  __shared__ __align__(16) _Float16 X[48 * KPAD];
  __shared__ float hbuf[48 * 128];
  int tid = threadIdx.x, n0 = blockIdx.x * 4;
  fill_X(X, af16, nbr_fea, nbr_idx, n0, tid);
  __syncthreads();

  int lane = tid & 31, wave = tid >> 5;
  v8f acc[16]; v8f zero = {};
  #pragma unroll
  for (int t = 0; t < 16; ++t) acc[t] = zero;
  gemm48(X, wp, lane, wave, acc);

  int col16 = lane & 15, hh = lane >> 4;
  #pragma unroll
  for (int t = 0; t < 8; ++t) {                    // filter tile t, core tile t+8
    int colF = t * 16 + col16;
    float sF = sc[colF],       fF = sh[colF];
    float sC = sc[colF + 128], fC = sh[colF + 128];
    #pragma unroll
    for (int j2 = 0; j2 < 8; ++j2) {
      float gf = acc[t][j2]     * sF + fF;
      float gc = acc[t + 8][j2] * sC + fC;
      int r48 = wave * 16 + j2 + 8 * hh;           // C-layout row
      hbuf[r48 * 128 + colF] = sigmoidf_(gf) * leakyf(gc);
    }
  }
  __syncthreads();
  for (int o = tid; o < 512; o += 96) {            // 4 atoms x 128 cols
    int a = o >> 7, col = o & 127;
    float s = 0.f;
    #pragma unroll
    for (int m = 0; m < MNBR; ++m) s += hbuf[(a * MNBR + m) * 128 + col];
    S[(size_t)(n0 + a) * 128 + col] = s;           // direct store, no atomics
  }
}

// ------------------------------- BN2 stats over N atoms (block per column)
__global__ void k_bn2_stats(const float* __restrict__ S,
                            const float* __restrict__ g2, const float* __restrict__ b2,
                            float* __restrict__ sc2, float* __restrict__ sh2) {
  __shared__ float rs[256], rq[256];
  int col = blockIdx.x, tid = threadIdx.x;
  float s = 0.f, q = 0.f;
  for (int n = tid; n < N_ATOM; n += 256) {
    float v = S[(size_t)n * 128 + col];
    s += v; q += v * v;
  }
  rs[tid] = s; rq[tid] = q;
  __syncthreads();
  for (int d = 128; d > 0; d >>= 1) {
    if (tid < d) { rs[tid] += rs[tid + d]; rq[tid] += rq[tid + d]; }
    __syncthreads();
  }
  if (tid == 0) {
    float mean = rs[0] / (float)N_ATOM;
    float var  = rq[0] / (float)N_ATOM - mean * mean;
    float rstd = rsqrtf(var + EPS_BN);
    float sv = g2[col] * rstd;
    sc2[col] = sv;
    sh2[col] = b2[col] - mean * sv;
  }
}

// ------------------------------- residual update: af = leaky(af + bn2(S))
__global__ void k_update_af(float* __restrict__ af, _Float16* __restrict__ af16,
                            const float* __restrict__ S,
                            const float* __restrict__ sc2, const float* __restrict__ sh2) {
  size_t g = (size_t)blockIdx.x * 256 + threadIdx.x;
  int c = (int)(g & 127);
  float v = af[g] + S[g] * sc2[c] + sh2[c];
  v = leakyf(v);
  af[g]   = v;
  af16[g] = (_Float16)v;
}

// ------------------------------- heads pass 2: BN + act + 128-dot + final act
__global__ __launch_bounds__(96) void k_heads_apply(
    const _Float16* __restrict__ af16, const float* __restrict__ nbr_fea,
    const int* __restrict__ nbr_idx, const _Float16* __restrict__ wp,
    const float* __restrict__ sc, const float* __restrict__ sh,
    const float* __restrict__ dist, const float* __restrict__ w2d,
    const float* __restrict__ b2d, const float* __restrict__ w2c,
    const float* __restrict__ b2c, float* __restrict__ out) {
  __shared__ __align__(16) _Float16 X[48 * KPAD];
  __shared__ _Float16 dbuf[48 * 128];
  __shared__ _Float16 cbuf[48 * 128];
  int tid = threadIdx.x, n0 = blockIdx.x * 4;
  fill_X(X, af16, nbr_fea, nbr_idx, n0, tid);
  __syncthreads();

  int lane = tid & 31, wave = tid >> 5;
  v8f acc[16]; v8f zero = {};
  #pragma unroll
  for (int t = 0; t < 16; ++t) acc[t] = zero;
  gemm48(X, wp, lane, wave, acc);

  int col16 = lane & 15, hh = lane >> 4;
  #pragma unroll
  for (int t = 0; t < 8; ++t) {                    // dist tiles 0..7, const 8..15
    int col = t * 16 + col16;
    float sD = sc[col],       fD = sh[col];
    float sC = sc[col + 128], fC = sh[col + 128];
    #pragma unroll
    for (int j2 = 0; j2 < 8; ++j2) {
      int r48 = wave * 16 + j2 + 8 * hh;
      dbuf[r48 * 128 + col] = (_Float16)leakyf(acc[t][j2] * sD + fD);
      cbuf[r48 * 128 + col] = (_Float16)softplusf_(acc[t + 8][j2] * sC + fC);
    }
  }
  __syncthreads();
  int r = tid >> 1, head = tid & 1;                // 48 rows x 2 heads = 96 threads
  size_t R = (size_t)n0 * MNBR + r;
  float dot = 0.f;
  if (head == 0) {
    #pragma unroll 8
    for (int c = 0; c < 128; ++c) dot += (float)dbuf[r * 128 + c] * w2d[c];
    out[R * 2 + 0] = leakyf(dot + b2d[0] + dist[R]);
  } else {
    #pragma unroll 8
    for (int c = 0; c < 128; ++c) dot += (float)cbuf[r * 128 + c] * w2c[c];
    out[R * 2 + 1] = softplusf_(dot + b2c[0]);
  }
}

// ------------------------------------------------------------------ host side
extern "C" void kernel_launch(void* const* d_in, const int* in_sizes, int n_in,
                              void* d_out, int out_size, void* d_ws, size_t ws_size,
                              hipStream_t stream) {
  (void)in_sizes; (void)n_in; (void)out_size; (void)ws_size;
  const float* atom_fea   = (const float*)d_in[0];
  const float* nbr_fea    = (const float*)d_in[1];
  const float* nbr_off    = (const float*)d_in[2];
  const float* atom_pos   = (const float*)d_in[3];
  const float* cells      = (const float*)d_in[4];
  const int*   nbr_idx    = (const int*)  d_in[5];
  const float* emb_W      = (const float*)d_in[6];
  const float* emb_b      = (const float*)d_in[7];
  const float* convs_W    = (const float*)d_in[8];
  /* d_in[9] convs_b: cancels in BN1 */
  const float* bn1_g      = (const float*)d_in[10];
  const float* bn1_b      = (const float*)d_in[11];
  const float* bn2_g      = (const float*)d_in[12];
  const float* bn2_b      = (const float*)d_in[13];
  const float* dist_W     = (const float*)d_in[14];
  /* d_in[15] dist_b: cancels in BN */
  const float* dist_bn_g  = (const float*)d_in[16];
  const float* dist_bn_b  = (const float*)d_in[17];
  const float* dist2_W    = (const float*)d_in[18];
  const float* dist2_b    = (const float*)d_in[19];
  const float* const_W    = (const float*)d_in[20];
  /* d_in[21] const_b: cancels in BN */
  const float* const_bn_g = (const float*)d_in[22];
  const float* const_bn_b = (const float*)d_in[23];
  const float* const2_W   = (const float*)d_in[24];
  const float* const2_b   = (const float*)d_in[25];
  float* out = (float*)d_out;

  // workspace carve-out (~27 MB total)
  char* w = (char*)d_ws;
  size_t off = 0;
  auto carve = [&](size_t bytes) -> void* {
    void* p = w + off;
    off = (off + bytes + 255) & ~(size_t)255;
    return p;
  };
  float*    af     = (float*)   carve((size_t)N_ATOM * 128 * 4);
  _Float16* af16   = (_Float16*)carve((size_t)N_ATOM * 128 * 2);
  float*    distb  = (float*)   carve((size_t)NROWS * 4);
  float*    S      = (float*)   carve((size_t)N_ATOM * 128 * 4);
  _Float16* wpack  = (_Float16*)carve((size_t)256 * KPAD * 2);
  float*    gstats = (float*)   carve(512 * 4);
  float*    sc1    = (float*)   carve(256 * 4);
  float*    sh1    = (float*)   carve(256 * 4);
  float*    sc2    = (float*)   carve(128 * 4);
  float*    sh2    = (float*)   carve(128 * 4);

  k_embed<<<N_ATOM, 128, 0, stream>>>(atom_fea, emb_W, emb_b, af, af16);
  k_distance<<<(NROWS + 255) / 256, 256, 0, stream>>>(atom_pos, nbr_off, cells, nbr_idx, distb);

  for (int i = 0; i < 3; ++i) {
    const float* Wi = convs_W + (size_t)i * 256 * IN2;
    k_wpack<<<320, 256, 0, stream>>>(Wi, Wi, 256, wpack);
    hipMemsetAsync(gstats, 0, 512 * 4, stream);
    k_gemm_stats<<<N_ATOM / 4, 96, 0, stream>>>(af16, nbr_fea, nbr_idx, wpack, gstats);
    k_finalize<<<1, 256, 0, stream>>>(gstats, (float)NROWS,
                                      bn1_g + i * 256, bn1_b + i * 256,
                                      bn1_g + i * 256, bn1_b + i * 256, 256, sc1, sh1);
    k_conv_apply<<<N_ATOM / 4, 96, 0, stream>>>(af16, nbr_fea, nbr_idx, wpack, sc1, sh1, S);
    k_bn2_stats<<<128, 256, 0, stream>>>(S, bn2_g + i * 128, bn2_b + i * 128, sc2, sh2);
    k_update_af<<<N_ATOM * 128 / 256, 256, 0, stream>>>(af, af16, S, sc2, sh2);
  }

  // bond heads: dist(128) || const(128) packed as one 256-column GEMM
  k_wpack<<<320, 256, 0, stream>>>(dist_W, const_W, 128, wpack);
  hipMemsetAsync(gstats, 0, 512 * 4, stream);
  k_gemm_stats<<<N_ATOM / 4, 96, 0, stream>>>(af16, nbr_fea, nbr_idx, wpack, gstats);
  k_finalize<<<1, 256, 0, stream>>>(gstats, (float)NROWS,
                                    dist_bn_g, dist_bn_b, const_bn_g, const_bn_b, 128, sc1, sh1);
  k_heads_apply<<<N_ATOM / 4, 96, 0, stream>>>(af16, nbr_fea, nbr_idx, wpack, sc1, sh1,
                                               distb, dist2_W, dist2_b, const2_W, const2_b, out);
}
